// Model_2_65274912964665
// MI455X (gfx1250) — compile-verified, matrix operand order checked by reading, not run on hardware
//
#include <hip/hip_runtime.h>
#include <hip/hip_bf16.h>
#include <hip/hip_fp16.h>

typedef float    v2f  __attribute__((ext_vector_type(2)));
typedef float    v8f  __attribute__((ext_vector_type(8)));
typedef _Float16 v8h  __attribute__((ext_vector_type(8)));
typedef _Float16 v16h __attribute__((ext_vector_type(16)));

#define N_NODES 8192
#define D_IN    512
#define D_H     256
#define D_Z     128

// ---------------------------------------------------------------------------
// zero-fill (atomic accumulators must be cleared every call)
// ---------------------------------------------------------------------------
__global__ void fill_zero_kernel(float* __restrict__ p, int n) {
  int i = blockIdx.x * blockDim.x + threadIdx.x;
  if (i < n) p[i] = 0.0f;
}

// ---------------------------------------------------------------------------
// C[M,N] = act(A)[M,K] @ B[K,N], row-major, full f32 precision via
// V_WMMA_F32_16X16X4_F32.  One wave -> one 16x16 C tile.
// A 16x4 f32 layout: lanes 0-15 hold {K0,K1} of row M=lane,
//                    lanes 16-31 hold {K2,K3} of row M=lane-16.
// B 4x16 f32 layout mirrors A with N=lane&15.
// C/D: 8 VGPRs, row M = r + 8*(lane>=16), col N = lane&15.
// K, N, RELU are compile-time: shifts instead of div, no per-element cndmask.
// ---------------------------------------------------------------------------
template <int K, int N, bool RELU>
__global__ void gemm_f32_wmma(const float* __restrict__ A,
                              const float* __restrict__ B,
                              float* __restrict__ C) {
  constexpr int NT = N / 16;
  const int wave  = (int)((blockIdx.x * blockDim.x + threadIdx.x) >> 5);
  const int lane  = threadIdx.x & 31;
  const int mTile = wave / NT;       // NT is a power of two -> shift
  const int nTile = wave % NT;
  const int half  = lane >> 4;       // which K-pair this lane provides
  const int l15   = lane & 15;

  const float* aRow = A + (size_t)(mTile * 16 + l15) * K;
  const float* bCol = B + (size_t)(nTile * 16 + l15);

  v8f acc = {};
#pragma unroll 4
  for (int k = 0; k < K; k += 4) {
    const int kk = k + 2 * half;
    v2f a = *(const v2f*)(aRow + kk);
    if (RELU) { a.x = fmaxf(a.x, 0.0f); a.y = fmaxf(a.y, 0.0f); }
    v2f b;
    b.x = bCol[(size_t)kk * N];
    b.y = bCol[(size_t)(kk + 1) * N];
    acc = __builtin_amdgcn_wmma_f32_16x16x4_f32(
        /*neg_a=*/false, a, /*neg_b=*/false, b,
        /*c_mod=*/(short)0, acc, /*reuse_a=*/false, /*reuse_b=*/false);
  }

  float* cOut = C + (size_t)(nTile * 16 + l15);
  const int rowBase = mTile * 16 + half * 8;
#pragma unroll
  for (int r = 0; r < 8; ++r)
    cOut[(size_t)(rowBase + r) * N] = acc[r];
}

// ---------------------------------------------------------------------------
// COO SPMM scatter-add: dst[rows[e], :] += vals[e] * src[cols[e], :]
// One thread = one edge x 4 features.  Destination is L2-resident (<=8MB).
// F compile-time -> all index math is shifts/masks, no 64-bit mul/div.
// ---------------------------------------------------------------------------
template <int F>
__global__ void spmm_atomic(const float* __restrict__ vals,
                            const int*   __restrict__ rows,
                            const int*   __restrict__ cols,
                            const float* __restrict__ src,
                            float*       __restrict__ dst,
                            int nEdges) {
  constexpr int CHUNKS = F / 4;              // power of two
  const long gid = (long)blockIdx.x * blockDim.x + threadIdx.x;
  if (gid >= (long)nEdges * CHUNKS) return;
  const int e = (int)(gid / CHUNKS);
  const int c = ((int)gid & (CHUNKS - 1)) * 4;

  const float  v = vals[e];
  const float4 s = *(const float4*)(src + (size_t)cols[e] * F + c);
  float*       d = dst + (size_t)rows[e] * F + c;
  unsafeAtomicAdd(d + 0, v * s.x);
  unsafeAtomicAdd(d + 1, v * s.y);
  unsafeAtomicAdd(d + 2, v * s.z);
  unsafeAtomicAdd(d + 3, v * s.w);
}

// ---------------------------------------------------------------------------
// z = relu(agg2); emit f32 (output 0) and f16 copy for the decoder GEMM
// ---------------------------------------------------------------------------
__global__ void finalize_z(const float* __restrict__ agg,
                           float* __restrict__ zf32,
                           _Float16* __restrict__ zf16, int n) {
  int i = blockIdx.x * blockDim.x + threadIdx.x;
  if (i >= n) return;
  float v = fmaxf(agg[i], 0.0f);
  zf32[i] = v;
  zf16[i] = (_Float16)v;
}

// ---------------------------------------------------------------------------
// rec = z @ z^T via V_WMMA_F32_16X16X32_F16, stored to both views.
// For B = z^T, the per-lane B layout equals A's layout of row n of z,
// so one loader serves both operands.
// f16 A 16x32 chunk layout: lanes 0-15: halves 0-7 = K+0..7, 8-15 = K+16..23;
//                           lanes 16-31: +8 on both ranges.
// ---------------------------------------------------------------------------
__device__ inline v16h load_zrow_chunk(const _Float16* __restrict__ rowPtr,
                                       int kc, int half) {
  const _Float16* p = rowPtr + kc * 32 + half * 8;
  v8h lo = *(const v8h*)(p);
  v8h hi = *(const v8h*)(p + 16);
  v16h r;
#pragma unroll
  for (int i = 0; i < 8; ++i) { r[i] = lo[i]; r[i + 8] = hi[i]; }
  return r;
}

__global__ void decoder_wmma(const _Float16* __restrict__ z16,
                             float* __restrict__ rec0,
                             float* __restrict__ rec1) {
  const int wave = (int)((blockIdx.x * blockDim.x + threadIdx.x) >> 5);
  const int lane = threadIdx.x & 31;
  const int mTile = wave >> 7;          // 512 M-tiles
  const int nBlk  = wave & 127;         // 128 N-blocks of 64
  const int half  = lane >> 4;
  const int l15   = lane & 15;
  const int mBase = mTile * 16;
  const int nBase = nBlk * 64;

  // A chunks (16 rows x 128 K) held in registers, reused over 4 N-subtiles
  const _Float16* aRow = z16 + (size_t)(mBase + l15) * D_Z;
  v16h a[4];
#pragma unroll
  for (int kc = 0; kc < 4; ++kc) a[kc] = load_zrow_chunk(aRow, kc, half);

#pragma unroll
  for (int ns = 0; ns < 4; ++ns) {
    const int col = nBase + ns * 16 + l15;
    const _Float16* bRow = z16 + (size_t)col * D_Z;
    v8f acc = {};
#pragma unroll
    for (int kc = 0; kc < 4; ++kc) {
      v16h b = load_zrow_chunk(bRow, kc, half);
      acc = __builtin_amdgcn_wmma_f32_16x16x32_f16(
          false, a[kc], false, b, (short)0, acc, false, false);
    }
#pragma unroll
    for (int r = 0; r < 8; ++r) {
      const size_t idx = (size_t)(mBase + half * 8 + r) * N_NODES + col;
      const float v = acc[r];
      rec0[idx] = v;   // view 0
      rec1[idx] = v;   // view 1 (identical by construction)
    }
  }
}

// ---------------------------------------------------------------------------
extern "C" void kernel_launch(void* const* d_in, const int* in_sizes, int n_in,
                              void* d_out, int out_size, void* d_ws, size_t ws_size,
                              hipStream_t stream) {
  const float* x     = (const float*)d_in[0];   // [8192,512]
  const float* W1    = (const float*)d_in[1];   // [512,256]
  const float* W2    = (const float*)d_in[2];   // [256,128]
  const float* evals = (const float*)d_in[3];   // [262144]
  const int*   erows = (const int*)d_in[4];
  const int*   ecols = (const int*)d_in[5];
  const int nEdges = in_sizes[3];

  float* out  = (float*)d_out;
  float* zOut = out;                                        // [8192,128]
  float* rec0 = out + (size_t)N_NODES * D_Z;                // [8192,8192]
  float* rec1 = rec0 + (size_t)N_NODES * N_NODES;           // view 1

  char* ws = (char*)d_ws;
  float*    support1 = (float*)(ws);                        // 8 MB
  float*    hbuf     = (float*)(ws + (size_t)(8  << 20));   // 8 MB (atomic acc)
  float*    s2       = (float*)(ws + (size_t)(16 << 20));   // 4 MB
  float*    agg2     = (float*)(ws + (size_t)(20 << 20));   // 4 MB (atomic acc)
  _Float16* z16      = (_Float16*)(ws + (size_t)(24 << 20));// 2 MB

  // clear atomic accumulators (fresh every call — no cross-call state)
  fill_zero_kernel<<<(N_NODES * D_H + 255) / 256, 256, 0, stream>>>(hbuf, N_NODES * D_H);
  fill_zero_kernel<<<(N_NODES * D_Z + 255) / 256, 256, 0, stream>>>(agg2, N_NODES * D_Z);

  // layer 1: support1 = x @ W1   (512x16 tiles = 8192 waves)
  {
    int waves = (N_NODES / 16) * (D_H / 16);
    gemm_f32_wmma<D_IN, D_H, false>
        <<<waves / 8, 256, 0, stream>>>(x, W1, support1);
  }
  // spmm1: hbuf = A @ support1
  {
    long thr = (long)nEdges * (D_H / 4);
    spmm_atomic<D_H><<<(int)((thr + 255) / 256), 256, 0, stream>>>(
        evals, erows, ecols, support1, hbuf, nEdges);
  }
  // layer 2: s2 = relu(hbuf) @ W2  (relu fused into A load, compile-time)
  {
    int waves = (N_NODES / 16) * (D_Z / 16);
    gemm_f32_wmma<D_H, D_Z, true>
        <<<waves / 8, 256, 0, stream>>>(hbuf, W2, s2);
  }
  // spmm2: agg2 = A @ s2
  {
    long thr = (long)nEdges * (D_Z / 4);
    spmm_atomic<D_Z><<<(int)((thr + 255) / 256), 256, 0, stream>>>(
        evals, erows, ecols, s2, agg2, nEdges);
  }
  // z = relu(agg2) -> output + f16 copy
  finalize_z<<<(N_NODES * D_Z + 255) / 256, 256, 0, stream>>>(
      agg2, zOut, z16, N_NODES * D_Z);

  // decoder: rec = z @ z^T, written to both views (store-bandwidth bound)
  {
    int waves = (N_NODES / 16) * (N_NODES / 64);   // 65536 waves
    decoder_wmma<<<waves / 8, 256, 0, stream>>>(z16, rec0, rec1);
  }
}
